// Qwen3VisionModel_2980707304201
// MI455X (gfx1250) — matrix operand order repaired and checked by reading, not compiled, not run on hardware
//
#include <hip/hip_runtime.h>
#include <hip/hip_bf16.h>
#include <cstdint>
#include <cstddef>

// ---------------------------------------------------------------------------
// Qwen3-Vision block for MI455X (gfx1250): f16 WMMA GEMMs + flash attention.
// ---------------------------------------------------------------------------

typedef _Float16 v16h __attribute__((ext_vector_type(16)));
typedef _Float16 v8h  __attribute__((ext_vector_type(8)));
typedef float    v8f  __attribute__((ext_vector_type(8)));

static constexpr int S_TOK = 16384;
static constexpr int DMODEL = 1152;
static constexpr int NHEAD = 16;
static constexpr int HD = 72;          // head dim
static constexpr int HDP = 96;         // padded head dim (multiple of 32)
static constexpr int IFF = 4304;       // ffn inner
static constexpr int IFFP = 4320;      // padded to multiple of 32
static constexpr int DQKV = 3 * DMODEL;

// -------------------------- WMMA fragment helpers --------------------------

__device__ __forceinline__ v8f wmma_f16(v16h a, v16h b, v8f c) {
  // D = A(16x32 f16) * B(32x16 f16) + C(16x16 f32)
  return __builtin_amdgcn_wmma_f32_16x16x32_f16(
      /*neg_a=*/false, a, /*neg_b=*/false, b,
      /*c_mod=*/(short)0, c, /*reuse_a=*/false, /*reuse_b=*/false);
}

// A fragment: lane m holds row m; elements 0..7 -> K = g*8+0..7,
// elements 8..15 -> K = g*8+16..23  (per ISA 16-bit A 16x32 layout).
__device__ __forceinline__ v16h load_frag_a(const _Float16* rowk0, int g) {
  v16h f;
  *(v8h*)&f       = *(const v8h*)(rowk0 + g * 8);
  *((v8h*)&f + 1) = *(const v8h*)(rowk0 + 16 + g * 8);
  return f;
}

// B fragment: lane n holds column n; elements 0..7 -> K = g*8+0..7,
// elements 8..15 -> K = g*8+16..23 (same as A).
__device__ __forceinline__ v16h load_frag_b(const _Float16* rowk0, int g) {
  v16h f;
  *(v8h*)&f       = *(const v8h*)(rowk0 + g * 8);
  *((v8h*)&f + 1) = *(const v8h*)(rowk0 + 16 + g * 8);
  return f;
}

// ------------------------------ small kernels -------------------------------

// f32 -> f16 weight convert with K padding: dst is [N, Kp], zeros for k >= K.
__global__ void cvt_w_f16(const float* __restrict__ src, _Float16* __restrict__ dst,
                          int N, int K, int Kp) {
  size_t tot = (size_t)N * Kp;
  for (size_t i = (size_t)blockIdx.x * blockDim.x + threadIdx.x; i < tot;
       i += (size_t)gridDim.x * blockDim.x) {
    size_t n = i / Kp;
    int k = (int)(i - n * Kp);
    dst[i] = (k < K) ? (_Float16)src[n * K + k] : (_Float16)0.0f;
  }
}

// Row layernorm: f32 in -> f16 out (feeds WMMA A operand).
__global__ __launch_bounds__(256) void layernorm_f16(
    const float* __restrict__ x, const float* __restrict__ gw,
    const float* __restrict__ bw, _Float16* __restrict__ out) {
  __shared__ float red[256];
  const int row = blockIdx.x, tid = threadIdx.x;
  const float* xr = x + (size_t)row * DMODEL;
  float s = 0.f, s2 = 0.f;
  for (int t = tid; t < DMODEL; t += 256) { float v = xr[t]; s += v; s2 += v * v; }
  red[tid] = s; __syncthreads();
  for (int st = 128; st > 0; st >>= 1) { if (tid < st) red[tid] += red[tid + st]; __syncthreads(); }
  const float mean = red[0] * (1.0f / DMODEL);
  __syncthreads();
  red[tid] = s2; __syncthreads();
  for (int st = 128; st > 0; st >>= 1) { if (tid < st) red[tid] += red[tid + st]; __syncthreads(); }
  const float var = red[0] * (1.0f / DMODEL) - mean * mean;
  const float rstd = rsqrtf(var + 1e-6f);
  _Float16* orow = out + (size_t)row * DMODEL;
  for (int t = tid; t < DMODEL; t += 256)
    orow[t] = (_Float16)((xr[t] - mean) * rstd * gw[t] + bw[t]);
}

// RoPE on q/k + f16 repack: q,k -> [H, S, HDP] (A/B friendly), v -> [H, HDP, S]
// (transposed so P@V B-fragments are contiguous). Pad cols 72..95 with zeros.
__global__ __launch_bounds__(HDP) void rope_pack(
    const float* __restrict__ qkv, const float* __restrict__ cosb,
    const float* __restrict__ sinb, _Float16* __restrict__ qp,
    _Float16* __restrict__ kp, _Float16* __restrict__ vt) {
  const int s = blockIdx.x, h = blockIdx.y, t = threadIdx.x;  // t in [0, 96)
  const size_t base = (size_t)s * DQKV + h * HD;
  _Float16 qo = 0, ko = 0, vo = 0;
  if (t < HD) {
    const float c = cosb[(size_t)s * HD + t];
    const float sn = sinb[(size_t)s * HD + t];
    const float q = qkv[base + t];
    const float k = qkv[base + DMODEL + t];
    const float v = qkv[base + 2 * DMODEL + t];
    const float qr = (t < HD / 2) ? -qkv[base + t + HD / 2] : qkv[base + t - HD / 2];
    const float kr = (t < HD / 2) ? -qkv[base + DMODEL + t + HD / 2]
                                  : qkv[base + DMODEL + t - HD / 2];
    qo = (_Float16)(q * c + qr * sn);
    ko = (_Float16)(k * c + kr * sn);
    vo = (_Float16)v;
  }
  qp[((size_t)h * S_TOK + s) * HDP + t] = qo;
  kp[((size_t)h * S_TOK + s) * HDP + t] = ko;
  vt[((size_t)h * HDP + t) * S_TOK + s] = vo;
}

// ------------------------------- WMMA GEMM ---------------------------------
// C[M,N] = A[M,K](f16) @ B[N,K](f16)^T, epilogues:
//   EPI 0: outF = acc + bias                       (f32)
//   EPI 1: outF = acc + bias + resid               (f32)
//   EPI 2: outH = gelu_tanh(acc + bias) f16, zero-pad cols [N, ldo)
// Block: 256 threads = 8 waves (2 x 4), wave tile 64x64 -> block 128x256.

template <int EPI>
__global__ __launch_bounds__(256, 1) void gemm_wmma(
    const _Float16* __restrict__ A, const _Float16* __restrict__ B,
    const float* __restrict__ bias, const float* __restrict__ resid,
    float* __restrict__ outF, _Float16* __restrict__ outH,
    int M, int N, int K, int ldo) {
  const int lane = threadIdx.x & 31;
  const int wid = threadIdx.x >> 5;
  const int wm = wid & 1, wn = wid >> 1;
  const int m0 = blockIdx.x * 128 + wm * 64;
  const int n0 = blockIdx.y * 256 + wn * 64;
  const int l15 = lane & 15, g = lane >> 4;

  const v8f fzero = {};
  const v16h hzero = {};
  v8f acc[4][4];
#pragma unroll
  for (int i = 0; i < 4; ++i)
#pragma unroll
    for (int j = 0; j < 4; ++j) acc[i][j] = fzero;

  const _Float16* arow[4];
#pragma unroll
  for (int i = 0; i < 4; ++i)
    arow[i] = A + (size_t)(m0 + i * 16 + l15) * K;

  const _Float16* brow[4];
  bool bok[4];
#pragma unroll
  for (int j = 0; j < 4; ++j) {
    const int n = n0 + j * 16 + l15;
    bok[j] = (n < N);
    brow[j] = B + (size_t)(bok[j] ? n : 0) * K;
  }

  for (int k0 = 0; k0 < K; k0 += 32) {
    v16h af[4], bf[4];
#pragma unroll
    for (int i = 0; i < 4; ++i) af[i] = load_frag_a(arow[i] + k0, g);
#pragma unroll
    for (int j = 0; j < 4; ++j) bf[j] = bok[j] ? load_frag_b(brow[j] + k0, g) : hzero;
#pragma unroll
    for (int i = 0; i < 4; ++i)
#pragma unroll
      for (int j = 0; j < 4; ++j) acc[i][j] = wmma_f16(af[i], bf[j], acc[i][j]);
  }

  // Epilogue. D layout: VGPR r -> row (m0 + i*16 + r + g*8), col n0 + j*16 + l15.
#pragma unroll
  for (int i = 0; i < 4; ++i) {
#pragma unroll
    for (int j = 0; j < 4; ++j) {
      const int col = n0 + j * 16 + l15;
#pragma unroll
      for (int r = 0; r < 8; ++r) {
        const int row = m0 + i * 16 + r + g * 8;
        if (col < N) {
          float v = acc[i][j][r] + bias[col];
          if (EPI == 1) v += resid[(size_t)row * N + col];
          if (EPI == 2) {
            const float u = v;
            const float gl = 0.5f * u *
                (1.0f + tanhf(0.7978845608028654f * (u + 0.044715f * u * u * u)));
            outH[(size_t)row * ldo + col] = (_Float16)gl;
          } else {
            outF[(size_t)row * N + col] = v;
          }
        } else if (EPI == 2 && col < ldo) {
          outH[(size_t)row * ldo + col] = (_Float16)0.0f;  // K-pad for FC2
        }
      }
    }
  }
}

// --------------------------- flash attention (WMMA) -------------------------
// grid (S/128, H); 8 waves/block, each wave owns a 16-row query tile.
// Key chunks of 32: S-tiles via 3x f16 WMMA over padded hd=96, online softmax
// with shfl_xor row reductions, P bounced through LDS (D-layout -> A-layout),
// P@V via 5 WMMAs against transposed V.

__global__ __launch_bounds__(256, 1) void attn_wmma(
    const _Float16* __restrict__ qp, const _Float16* __restrict__ kp,
    const _Float16* __restrict__ vt, _Float16* __restrict__ attn,
    const int* __restrict__ cu) {
  __shared__ alignas(16) _Float16 plds[8][16 * 32];
  const int lane = threadIdx.x & 31;
  const int wid = threadIdx.x >> 5;
  const int l15 = lane & 15, g = lane >> 4;
  const int h = blockIdx.y;
  const int m0 = blockIdx.x * 128 + wid * 16;

  // Locate sequence [a,b) containing this (128-aligned) query tile.
  int a = 0, b = S_TOK;
  for (int i = 0; i < 8; ++i) {
    const int lo = cu[i], hi = cu[i + 1];
    if (m0 >= lo && m0 < hi) { a = lo; b = hi; }
  }

  const v8f fzero = {};
  v8f o[5];
#pragma unroll
  for (int t = 0; t < 5; ++t) o[t] = fzero;
  float mrow[8], lrow[8];
#pragma unroll
  for (int r = 0; r < 8; ++r) { mrow[r] = -3.0e38f; lrow[r] = 0.0f; }

  const float scale = rsqrtf((float)HD);
  const _Float16* qrow = qp + ((size_t)h * S_TOK + m0 + l15) * HDP;
  const _Float16* kbase = kp + (size_t)h * S_TOK * HDP;
  const _Float16* vbase = vt + (size_t)h * HDP * S_TOK;

  for (int kc = a; kc < b; kc += 32) {
    // ---- S = Q K^T over padded head dim (zeros beyond 72 contribute 0) ----
    v8f s0 = fzero, s1 = fzero;
#pragma unroll
    for (int kh = 0; kh < HDP; kh += 32) {
      const v16h aq = load_frag_a(qrow + kh, g);
      const v16h bk0 = load_frag_b(kbase + (size_t)(kc + l15) * HDP + kh, g);
      const v16h bk1 = load_frag_b(kbase + (size_t)(kc + 16 + l15) * HDP + kh, g);
      s0 = wmma_f16(aq, bk0, s0);
      s1 = wmma_f16(aq, bk1, s1);
    }

    // ---- online softmax update (row r lives on VGPR r, 16 lanes wide) ----
#pragma unroll
    for (int r = 0; r < 8; ++r) {
      const float x0 = s0[r] * scale, x1 = s1[r] * scale;
      float mx = fmaxf(x0, x1);
#pragma unroll
      for (int off = 1; off < 16; off <<= 1) mx = fmaxf(mx, __shfl_xor(mx, off, 16));
      const float mnew = fmaxf(mrow[r], mx);
      const float alpha = __expf(mrow[r] - mnew);
      const float p0 = __expf(x0 - mnew), p1 = __expf(x1 - mnew);
      float rs = p0 + p1;
#pragma unroll
      for (int off = 1; off < 16; off <<= 1) rs += __shfl_xor(rs, off, 16);
      lrow[r] = lrow[r] * alpha + rs;
      mrow[r] = mnew;
#pragma unroll
      for (int t = 0; t < 5; ++t) o[t][r] *= alpha;
      const int row = r + g * 8;
      plds[wid][row * 32 + l15] = (_Float16)p0;
      plds[wid][row * 32 + 16 + l15] = (_Float16)p1;
    }
    // LDS same-wave RAW: DS ops are in-order; make ordering explicit.
    asm volatile("s_wait_dscnt 0" ::: "memory");

    // ---- O += P V : P reloaded from LDS in A-layout, V^T B-fragments ----
    const v16h ap = load_frag_a(&plds[wid][l15 * 32], g);
#pragma unroll
    for (int t = 0; t < 5; ++t) {
      const v16h bv = load_frag_b(vbase + (size_t)(t * 16 + l15) * S_TOK + kc, g);
      o[t] = wmma_f16(ap, bv, o[t]);
    }
  }

  // ---- epilogue: O /= l, store f16 [S, H*72] ----
#pragma unroll
  for (int t = 0; t < 5; ++t) {
    const int col = t * 16 + l15;
    if (col < HD) {
#pragma unroll
      for (int r = 0; r < 8; ++r) {
        const int row = m0 + r + g * 8;
        attn[(size_t)row * DMODEL + h * HD + col] = (_Float16)(o[t][r] / lrow[r]);
      }
    }
  }
}

// ------------------------------- launch glue --------------------------------

extern "C" void kernel_launch(void* const* d_in, const int* in_sizes, int n_in,
                              void* d_out, int out_size, void* d_ws, size_t ws_size,
                              hipStream_t stream) {
  const float* x     = (const float*)d_in[0];
  const float* cosb  = (const float*)d_in[1];
  const float* sinb  = (const float*)d_in[2];
  const float* ln1g  = (const float*)d_in[3];
  const float* ln1b  = (const float*)d_in[4];
  const float* ln2g  = (const float*)d_in[5];
  const float* ln2b  = (const float*)d_in[6];
  const float* qkvw  = (const float*)d_in[7];
  const float* qkvb  = (const float*)d_in[8];
  const float* projw = (const float*)d_in[9];
  const float* projb = (const float*)d_in[10];
  const float* fc1w  = (const float*)d_in[11];
  const float* fc1b  = (const float*)d_in[12];
  const float* fc2w  = (const float*)d_in[13];
  const float* fc2b  = (const float*)d_in[14];
  const int*   cu    = (const int*)d_in[15];
  float* out = (float*)d_out;
  char* ws = (char*)d_ws;

  // ---- workspace layout (bytes); qkv f32 region is reused after RoPE ----
  size_t off = 0;
  auto take = [&](size_t bytes) { size_t o = off; off += (bytes + 255) & ~(size_t)255; return o; };
  _Float16* wqkv = (_Float16*)(ws + take((size_t)DQKV * DMODEL * 2));
  _Float16* wprj = (_Float16*)(ws + take((size_t)DMODEL * DMODEL * 2));
  _Float16* wfc1 = (_Float16*)(ws + take((size_t)IFF * DMODEL * 2));
  _Float16* wfc2 = (_Float16*)(ws + take((size_t)DMODEL * IFFP * 2));
  _Float16* h16  = (_Float16*)(ws + take((size_t)S_TOK * DMODEL * 2));
  const size_t oR = take((size_t)S_TOK * DQKV * 4);          // qkv f32 region
  float*    qkvf  = (float*)(ws + oR);
  _Float16* attnh = (_Float16*)(ws + oR);                                  // reuse
  _Float16* h2    = (_Float16*)(ws + oR + (size_t)S_TOK * DMODEL * 2);     // reuse
  _Float16* gbuf  = (_Float16*)(ws + oR + (size_t)S_TOK * DMODEL * 4);     // reuse
  _Float16* qp = (_Float16*)(ws + take((size_t)NHEAD * S_TOK * HDP * 2));
  _Float16* kp = (_Float16*)(ws + take((size_t)NHEAD * S_TOK * HDP * 2));
  _Float16* vt = (_Float16*)(ws + take((size_t)NHEAD * HDP * S_TOK * 2));
  float*    x2 = (float*)(ws + take((size_t)S_TOK * DMODEL * 4));
  (void)ws_size; (void)in_sizes; (void)n_in; (void)out_size;

  // 1) weights -> f16 (FC2 padded K: 4304 -> 4320)
  cvt_w_f16<<<1024, 256, 0, stream>>>(qkvw, wqkv, DQKV, DMODEL, DMODEL);
  cvt_w_f16<<<1024, 256, 0, stream>>>(projw, wprj, DMODEL, DMODEL, DMODEL);
  cvt_w_f16<<<1024, 256, 0, stream>>>(fc1w, wfc1, IFF, DMODEL, DMODEL);
  cvt_w_f16<<<1024, 256, 0, stream>>>(fc2w, wfc2, DMODEL, IFF, IFFP);

  // 2) LN1
  layernorm_f16<<<S_TOK, 256, 0, stream>>>(x, ln1g, ln1b, h16);

  // 3) QKV GEMM: [S,1152] x [3456,1152]^T + b -> f32
  gemm_wmma<0><<<dim3(S_TOK / 128, (DQKV + 255) / 256), 256, 0, stream>>>(
      h16, wqkv, qkvb, nullptr, qkvf, nullptr, S_TOK, DQKV, DMODEL, 0);

  // 4) RoPE + f16 repack (q,k -> [H,S,96]; v -> [H,96,S] transposed)
  rope_pack<<<dim3(S_TOK, NHEAD), HDP, 0, stream>>>(qkvf, cosb, sinb, qp, kp, vt);

  // 5) varlen flash attention -> attn f16 [S, D]
  attn_wmma<<<dim3(S_TOK / 128, NHEAD), 256, 0, stream>>>(qp, kp, vt, attnh, cu);

  // 6) proj GEMM + residual: x2 = x + attn @ Wp^T + b (f32)
  gemm_wmma<1><<<dim3(S_TOK / 128, (DMODEL + 255) / 256), 256, 0, stream>>>(
      attnh, wprj, projb, x, x2, nullptr, S_TOK, DMODEL, DMODEL, 0);

  // 7) LN2
  layernorm_f16<<<S_TOK, 256, 0, stream>>>(x2, ln2g, ln2b, h2);

  // 8) FC1 + GELU -> f16 [S, 4320] (zero K-pad)
  gemm_wmma<2><<<dim3(S_TOK / 128, (IFF + 255) / 256), 256, 0, stream>>>(
      h2, wfc1, fc1b, nullptr, nullptr, gbuf, S_TOK, IFF, DMODEL, IFFP);

  // 9) FC2 + residual -> d_out (f32)
  gemm_wmma<1><<<dim3(S_TOK / 128, (DMODEL + 255) / 256), 256, 0, stream>>>(
      gbuf, wfc2, fc2b, x2, out, nullptr, S_TOK, DMODEL, IFFP, 0);
}
